// GATLayer_64991445123247
// MI455X (gfx1250) — matrix operand order repaired and checked by reading, not compiled, not run on hardware
//
#include <hip/hip_runtime.h>
#include <math.h>

typedef __attribute__((ext_vector_type(16))) __bf16 v16bf;
typedef __attribute__((ext_vector_type(8)))  float  v8f;

#define DIN 128

// ---------------------------------------------------------------------------
// helpers
// ---------------------------------------------------------------------------
__device__ __forceinline__ float warp_sum(float v) {
    #pragma unroll
    for (int off = 16; off; off >>= 1) v += __shfl_xor(v, off, 32);
    return v;
}

__device__ __forceinline__ float atomicMaxFloat(float* addr, float val) {
    int iv = __float_as_int(val);
    int old;
    if (iv >= 0) old = atomicMax((int*)addr, iv);
    else         old = (int)atomicMin((unsigned int*)addr, (unsigned int)iv);
    return __int_as_float(old);
}

__device__ __forceinline__ unsigned short bfbits(__bf16 h) {
    return __builtin_bit_cast(unsigned short, h);
}

__device__ __forceinline__ unsigned pack_bf16_pair(float e0, float e1) {
    return ((unsigned)bfbits((__bf16)e1) << 16) | (unsigned)bfbits((__bf16)e0);
}

// ---------------------------------------------------------------------------
// LayerNorm: one block (128 threads) per node/row
// ---------------------------------------------------------------------------
__global__ void ln_kernel(const float* __restrict__ in,
                          const float* __restrict__ g,
                          const float* __restrict__ b,
                          float* __restrict__ out) {
    int n = blockIdx.x, t = threadIdx.x;
    __shared__ float red[8];
    int wid = t >> 5, lane = t & 31;
    float v = in[(size_t)n * DIN + t];
    float s = warp_sum(v);
    if (lane == 0) red[wid] = s;
    __syncthreads();
    float mean = (red[0] + red[1] + red[2] + red[3]) * (1.0f / DIN);
    float d = v - mean;
    float s2 = warp_sum(d * d);
    if (lane == 0) red[4 + wid] = s2;
    __syncthreads();
    float var = (red[4] + red[5] + red[6] + red[7]) * (1.0f / DIN);
    out[(size_t)n * DIN + t] = d * rsqrtf(var + 1e-5f) * g[t] + b[t];
}

// residual + bias then LayerNorm: hres = agg + gat_b + x ; hn = LN(hres)
__global__ void finalize_kernel(const float* __restrict__ agg,
                                const float* __restrict__ x,
                                const float* __restrict__ gat_b,
                                const float* __restrict__ g,
                                const float* __restrict__ b,
                                float* __restrict__ hres,
                                float* __restrict__ hn) {
    int n = blockIdx.x, t = threadIdx.x;
    __shared__ float red[8];
    int wid = t >> 5, lane = t & 31;
    float v = agg[(size_t)n * DIN + t] + gat_b[t] + x[(size_t)n * DIN + t];
    hres[(size_t)n * DIN + t] = v;
    float s = warp_sum(v);
    if (lane == 0) red[wid] = s;
    __syncthreads();
    float mean = (red[0] + red[1] + red[2] + red[3]) * (1.0f / DIN);
    float d = v - mean;
    float s2 = warp_sum(d * d);
    if (lane == 0) red[4 + wid] = s2;
    __syncthreads();
    float var = (red[4] + red[5] + red[6] + red[7]) * (1.0f / DIN);
    hn[(size_t)n * DIN + t] = d * rsqrtf(var + 1e-5f) * g[t] + b[t];
}

// ---------------------------------------------------------------------------
// Pack fp32 weight B[K][NOUT] into bf16 hi/lo in WMMA B-operand lane layout.
// j-major within lane: index t = ((ct*NK + ks)*32 + lane)*8 + j
// so each lane's 8-dword fragment is 32 contiguous bytes (-> b128 loads).
// lane: n = ct*16 + (lane&15), K-half = (lane>>4)*16, dword j packs K=koff+2j,+1
// ---------------------------------------------------------------------------
__global__ void pack_b_kernel(const float* __restrict__ B,
                              unsigned* __restrict__ Bhi,
                              unsigned* __restrict__ Blo,
                              int K, int NOUT) {
    int t = blockIdx.x * blockDim.x + threadIdx.x;
    int NK = K >> 5;
    int total = (NOUT >> 4) * NK * 8 * 32;
    if (t >= total) return;
    int j    = t & 7;
    int lane = (t >> 3) & 31;
    int rest = t >> 8;
    int ks   = rest % NK;
    int ct   = rest / NK;
    int n    = ct * 16 + (lane & 15);
    int koff = (lane >> 4) * 16;
    int k    = ks * 32 + koff + 2 * j;
    float e0 = B[k * NOUT + n];
    float e1 = B[(k + 1) * NOUT + n];
    __bf16 h0 = (__bf16)e0, h1 = (__bf16)e1;
    Bhi[t] = ((unsigned)bfbits(h1) << 16) | (unsigned)bfbits(h0);
    Blo[t] = pack_bf16_pair(e0 - (float)h0, e1 - (float)h1);
}

// ---------------------------------------------------------------------------
// WMMA GEMM: out[N,NOUT] = act(A[N,K] @ B + bias) (+ residual)
// bf16x3 split-precision (hi*hi + lo*hi + hi*lo), fp32 accumulate.
// Stage: async-copy raw A tile (16 rows, contiguous) into LDS (ASYNCcnt),
//        cooperatively convert ONCE into packed WMMA A-layout hi/lo in LDS,
//        then each wave (one 16x16 tile) reads fragments via ds_load_b128
//        and B fragments via contiguous global b128 loads.
// ---------------------------------------------------------------------------
template <int K, int NOUT, int ACT, bool HASBIAS, bool HASRES>
__global__ __launch_bounds__(NOUT * 2) void gemm_bf16x3_kernel(
        const float* __restrict__ A,
        const unsigned* __restrict__ Bhi,
        const unsigned* __restrict__ Blo,
        const float* __restrict__ bias,
        const float* __restrict__ res,
        float* __restrict__ out, int N) {
    constexpr int NK  = K / 32;
    constexpr int NT  = NOUT * 2;        // threads per block
    __shared__ float    lds_rawf[16 * K];       // raw fp32 A tile
    __shared__ unsigned lds_ahi[NK * 256];      // packed bf16-hi A fragments
    __shared__ unsigned lds_alo[NK * 256];      // packed bf16-lo A fragments

    const int tid  = threadIdx.x;
    const int row0 = blockIdx.x * 16;

    // ---- phase 0: async linear copy of the 16xK fp32 tile into LDS --------
    {
        const unsigned long long abase =
            (unsigned long long)(uintptr_t)(A + (size_t)row0 * K);
        const unsigned lraw = (unsigned)(uintptr_t)(&lds_rawf[0]);
        #pragma unroll
        for (int c = tid; c < 4 * K; c += NT) {   // 16*K*4B / 16B chunks
            unsigned           loff = lraw + (unsigned)c * 16u;
            unsigned long long ga   = abase + (unsigned long long)c * 16ull;
            asm volatile("global_load_async_to_lds_b128 %0, %1, off"
                         :: "v"(loff), "v"(ga) : "memory");
        }
        asm volatile("s_wait_asynccnt 0x0" ::: "memory");
    }
    __syncthreads();

    // ---- phase 1: convert raw tile -> packed WMMA A-operand hi/lo ---------
    #pragma unroll
    for (int idx = tid; idx < NK * 256; idx += NT) {
        int j  = idx & 7;
        int l  = (idx >> 3) & 31;
        int ks = idx >> 8;
        int m = l & 15, khalf = (l >> 4) << 4;
        int i0 = 2 * j;                              // element pair index
        int kk = ks * 32 + khalf + i0;
        float2 e = *(const float2*)(lds_rawf + m * K + kk);
        __bf16 h0 = (__bf16)e.x, h1 = (__bf16)e.y;
        lds_ahi[idx] = ((unsigned)bfbits(h1) << 16) | (unsigned)bfbits(h0);
        lds_alo[idx] = pack_bf16_pair(e.x - (float)h0, e.y - (float)h1);
    }
    __syncthreads();

    // ---- phase 2: per-wave 16x16 tile, WMMA bf16x3 -------------------------
    const int ct   = tid >> 5;
    const int lane = tid & 31;
    v8f acc = {};

    #pragma unroll
    for (int ks = 0; ks < NK; ++ks) {
        union UV { uint4 q[2]; v16bf v; } ahi, alo, bh, bl;
        const uint4* pa = (const uint4*)(lds_ahi + ((ks << 5) + lane) * 8);
        ahi.q[0] = pa[0]; ahi.q[1] = pa[1];
        const uint4* pl = (const uint4*)(lds_alo + ((ks << 5) + lane) * 8);
        alo.q[0] = pl[0]; alo.q[1] = pl[1];
        const uint4* pb = (const uint4*)(Bhi + (size_t)(((ct * NK + ks) << 5) + lane) * 8);
        bh.q[0] = pb[0]; bh.q[1] = pb[1];
        const uint4* pq = (const uint4*)(Blo + (size_t)(((ct * NK + ks) << 5) + lane) * 8);
        bl.q[0] = pq[0]; bl.q[1] = pq[1];

        acc = __builtin_amdgcn_wmma_f32_16x16x32_bf16(false, ahi.v, false, bh.v,
                                                      (short)0, acc, false, false);
        acc = __builtin_amdgcn_wmma_f32_16x16x32_bf16(false, alo.v, false, bh.v,
                                                      (short)0, acc, false, false);
        acc = __builtin_amdgcn_wmma_f32_16x16x32_bf16(false, ahi.v, false, bl.v,
                                                      (short)0, acc, false, false);
    }

    const int col   = ct * 16 + (lane & 15);
    const int rbase = row0 + ((lane >> 4) << 3);
    const float bs  = HASBIAS ? bias[col] : 0.0f;
    #pragma unroll
    for (int r = 0; r < 8; ++r) {
        int row = rbase + r;
        if (row < N) {
            float v = acc[r] + bs;
            if (ACT == 1) v = 0.5f * v * (1.0f + erff(v * 0.70710678118654752f));
            if (HASRES) v += res[(size_t)row * NOUT + col];
            out[(size_t)row * NOUT + col] = v;
        }
    }
}

// ---------------------------------------------------------------------------
// Attention logits per node (+ init amax/denom): block 128 thr, warp per head
// ---------------------------------------------------------------------------
__global__ void att_logits_kernel(const float* __restrict__ xh,
                                  const float* __restrict__ att_src,
                                  const float* __restrict__ att_dst,
                                  float* __restrict__ a_src,
                                  float* __restrict__ a_dst,
                                  float* __restrict__ amax,
                                  float* __restrict__ denom) {
    int n = blockIdx.x, t = threadIdx.x;
    int head = t >> 5, lane = t & 31;
    float v  = xh[(size_t)n * DIN + t];
    float ps = warp_sum(v * att_src[t]);
    float pd = warp_sum(v * att_dst[t]);
    if (lane == 0) {
        a_src[n * 4 + head] = ps;
        a_dst[n * 4 + head] = pd;
        amax [n * 4 + head] = -INFINITY;
        denom[n * 4 + head] = 0.0f;
    }
}

__global__ void zero_kernel(float* __restrict__ p, int total) {
    int t = blockIdx.x * blockDim.x + threadIdx.x;
    if (t < total) p[t] = 0.0f;
}

// ---------------------------------------------------------------------------
// Segment softmax over destination nodes (3 passes), self-loops appended
// ---------------------------------------------------------------------------
__device__ __forceinline__ void edge_sd(const int* ei, int e, int E, int& s, int& d) {
    if (e < E) { s = ei[e]; d = ei[E + e]; }
    else       { s = d = e - E; }
}

__global__ void edge_max_kernel(const int* __restrict__ ei,
                                const float* __restrict__ a_src,
                                const float* __restrict__ a_dst,
                                float* __restrict__ amax, int E, int ET) {
    int t = blockIdx.x * blockDim.x + threadIdx.x;
    if (t >= ET * 4) return;
    int e = t >> 2, h = t & 3, s, d;
    edge_sd(ei, e, E, s, d);
    float v = a_src[s * 4 + h] + a_dst[d * 4 + h];
    v = v > 0.0f ? v : 0.2f * v;                      // leaky_relu
    atomicMaxFloat(&amax[d * 4 + h], v);
}

__global__ void edge_exp_kernel(const int* __restrict__ ei,
                                const float* __restrict__ a_src,
                                const float* __restrict__ a_dst,
                                const float* __restrict__ amax,
                                float* __restrict__ alpha,
                                float* __restrict__ denom, int E, int ET) {
    int t = blockIdx.x * blockDim.x + threadIdx.x;
    if (t >= ET * 4) return;
    int e = t >> 2, h = t & 3, s, d;
    edge_sd(ei, e, E, s, d);
    float v = a_src[s * 4 + h] + a_dst[d * 4 + h];
    v = v > 0.0f ? v : 0.2f * v;
    float a = expf(v - amax[d * 4 + h]);
    alpha[t] = a;
    atomicAdd(&denom[d * 4 + h], a);
}

// wave per edge: gather xh[src] row (float4/lane), scatter-add into agg[dst]
__global__ void edge_agg_kernel(const int* __restrict__ ei,
                                const float* __restrict__ xh,
                                const float* __restrict__ alpha,
                                const float* __restrict__ denom,
                                float* __restrict__ agg, int E, int ET) {
    int gid = blockIdx.x * blockDim.x + threadIdx.x;
    int e = gid >> 5, lane = gid & 31;
    if (e >= ET) return;
    int s, d;
    edge_sd(ei, e, E, s, d);
    int c0 = lane * 4;
    int head = lane >> 3;                              // c0/32
    float w = alpha[e * 4 + head] / denom[d * 4 + head];
    float4 v = *(const float4*)(xh + (size_t)s * DIN + c0);
    float* o = agg + (size_t)d * DIN + c0;
    atomicAdd(o + 0, v.x * w);
    atomicAdd(o + 1, v.y * w);
    atomicAdd(o + 2, v.z * w);
    atomicAdd(o + 3, v.w * w);
}

// ---------------------------------------------------------------------------
extern "C" void kernel_launch(void* const* d_in, const int* in_sizes, int n_in,
                              void* d_out, int out_size, void* d_ws, size_t ws_size,
                              hipStream_t stream) {
    const float* x       = (const float*)d_in[0];
    const int*   ei      = (const int*)  d_in[1];
    const float* ln1_g   = (const float*)d_in[2];
    const float* ln1_b   = (const float*)d_in[3];
    const float* W       = (const float*)d_in[4];
    const float* att_src = (const float*)d_in[5];
    const float* att_dst = (const float*)d_in[6];
    const float* gat_b   = (const float*)d_in[7];
    const float* ln2_g   = (const float*)d_in[8];
    const float* ln2_b   = (const float*)d_in[9];
    const float* w1      = (const float*)d_in[10];
    const float* b1      = (const float*)d_in[11];
    const float* w2      = (const float*)d_in[12];
    const float* b2      = (const float*)d_in[13];

    const int N  = in_sizes[0] / DIN;
    const int E  = in_sizes[1] / 2;
    const int ET = E + N;

    // workspace carve-up (fp32 unless noted)
    float* wsf   = (float*)d_ws;
    size_t o = 0;
    float* h_agg = wsf + o; o += (size_t)N * DIN;   // LN1 output, later reused as agg
    float* xh    = wsf + o; o += (size_t)N * DIN;   // GEMM1 output, later reused as hres
    float* hn    = wsf + o; o += (size_t)N * DIN;
    float* t1    = wsf + o; o += (size_t)N * 2 * DIN;
    float* a_src = wsf + o; o += (size_t)N * 4;
    float* a_dst = wsf + o; o += (size_t)N * 4;
    float* amax  = wsf + o; o += (size_t)N * 4;
    float* denom = wsf + o; o += (size_t)N * 4;
    float* alpha = wsf + o; o += (size_t)ET * 4;
    unsigned* Wp_hi  = (unsigned*)(wsf + o); o += 8192;   // 8ct*4ks*8*32
    unsigned* Wp_lo  = (unsigned*)(wsf + o); o += 8192;
    unsigned* w1p_hi = (unsigned*)(wsf + o); o += 16384;  // 16ct*4ks*8*32
    unsigned* w1p_lo = (unsigned*)(wsf + o); o += 16384;
    unsigned* w2p_hi = (unsigned*)(wsf + o); o += 16384;  // 8ct*8ks*8*32
    unsigned* w2p_lo = (unsigned*)(wsf + o); o += 16384;

    const int rowTiles = (N + 15) / 16;

    // 1) LN1
    ln_kernel<<<N, DIN, 0, stream>>>(x, ln1_g, ln1_b, h_agg);

    // 2) pack weights into WMMA B layout (bf16 hi/lo)
    pack_b_kernel<<<(8192  + 255) / 256, 256, 0, stream>>>(W,  Wp_hi,  Wp_lo,  128, 128);
    pack_b_kernel<<<(16384 + 255) / 256, 256, 0, stream>>>(w1, w1p_hi, w1p_lo, 128, 256);
    pack_b_kernel<<<(16384 + 255) / 256, 256, 0, stream>>>(w2, w2p_hi, w2p_lo, 256, 128);

    // 3) xh = h @ W   (WMMA bf16x3)
    gemm_bf16x3_kernel<128, 128, 0, false, false><<<rowTiles, 256, 0, stream>>>(
        h_agg, Wp_hi, Wp_lo, nullptr, nullptr, xh, N);

    // 4) per-node attention logits + init segment buffers
    att_logits_kernel<<<N, DIN, 0, stream>>>(xh, att_src, att_dst,
                                             a_src, a_dst, amax, denom);

    // 5) zero aggregation buffer (reuse h)
    zero_kernel<<<((N * DIN) + 255) / 256, 256, 0, stream>>>(h_agg, N * DIN);

    // 6-8) segment softmax + weighted scatter
    int eThreads = ET * 4;
    edge_max_kernel<<<(eThreads + 255) / 256, 256, 0, stream>>>(ei, a_src, a_dst,
                                                                amax, E, ET);
    edge_exp_kernel<<<(eThreads + 255) / 256, 256, 0, stream>>>(ei, a_src, a_dst,
                                                                amax, alpha, denom, E, ET);
    edge_agg_kernel<<<((ET * 32) + 255) / 256, 256, 0, stream>>>(ei, xh, alpha,
                                                                 denom, h_agg, E, ET);

    // 9) hres = agg + gat_b + x ; hn = LN2(hres)   (hres overwrites xh buffer)
    finalize_kernel<<<N, DIN, 0, stream>>>(h_agg, x, gat_b, ln2_g, ln2_b, xh, hn);

    // 10) t1 = gelu(hn @ w1 + b1)
    gemm_bf16x3_kernel<128, 256, 1, true, false><<<rowTiles, 512, 0, stream>>>(
        hn, w1p_hi, w1p_lo, b1, nullptr, t1, N);

    // 11) out = hres + (t1 @ w2 + b2)
    gemm_bf16x3_kernel<256, 128, 0, true, true><<<rowTiles, 256, 0, stream>>>(
        t1, w2p_hi, w2p_lo, b2, xh, (float*)d_out, N);
}
